// MyRGCNConv_37074157699596
// MI455X (gfx1250) — matrix operand order repaired and checked by reading, not compiled, not run on hardware
//
#include <hip/hip_runtime.h>
#include <hip/hip_bf16.h>

typedef __attribute__((ext_vector_type(16))) __bf16 v16bf;
typedef __attribute__((ext_vector_type(8)))  __bf16 v8bf;
typedef __attribute__((ext_vector_type(8)))  float  v8f;

__device__ __forceinline__ __bf16 f2bf(float f) {
    union { float f; unsigned u; } in; in.f = f;
    unsigned u = in.u;
    u += 0x7FFFu + ((u >> 16) & 1u);          // round-to-nearest-even
    unsigned short s = (unsigned short)(u >> 16);
    union { unsigned short s; __bf16 b; } out; out.s = s;
    return out.b;
}

// ---------------------------------------------------------------------------
// Kernel 1: convert linear[R][K=64][N=64] (fp32) -> Wt[R][N=64][K=64] (bf16)
// ---------------------------------------------------------------------------
__global__ void wt_bf16_kernel(const float* __restrict__ W,
                               __bf16* __restrict__ Wt, int total) {
    int t = blockIdx.x * blockDim.x + threadIdx.x;
    if (t >= total) return;
    int r   = t >> 12;        // / 4096
    int rem = t & 4095;
    int n   = rem >> 6;
    int k   = rem & 63;
    Wt[t] = f2bf(W[(r << 12) + (k << 6) + n]);
}

// ---------------------------------------------------------------------------
// Kernel 2: convert x (fp32) -> xb (bf16) once, so the 12x-per-node gather
// in the main kernel moves half the bytes and needs no VALU conversion.
// ---------------------------------------------------------------------------
__global__ void xb_bf16_kernel(const float* __restrict__ x,
                               __bf16* __restrict__ xb, int total) {
    int t = blockIdx.x * blockDim.x + threadIdx.x;
    if (t >= total) return;
    xb[t] = f2bf(x[t]);
}

// ---------------------------------------------------------------------------
// Kernel 3: seg[e] = searchsorted(ptr, e, 'right') - 1
// ---------------------------------------------------------------------------
__global__ void seg_kernel(const int* __restrict__ ptr, int num_node, int E,
                           int* __restrict__ seg) {
    int e = blockIdx.x * blockDim.x + threadIdx.x;
    if (e >= E) return;
    int lo = 0, hi = num_node;           // answer in [0, num_node-1]
    while (hi - lo > 1) {
        int mid = (lo + hi) >> 1;
        if (ptr[mid] <= e) lo = mid; else hi = mid;
    }
    seg[e] = lo;
}

// ---------------------------------------------------------------------------
// Kernel 4: WMMA per-edge transform + atomic scatter-sum.
// One wave32 handles a tile of 16 edges (M=16), N=64 output channels.
// All relation weights (<=64KB bf16) are staged into LDS once per block via
// the Tensor Data Mover; B operands then come from ds_load_b128.
// ---------------------------------------------------------------------------
template<bool USE_XB>
__global__ void __launch_bounds__(256)
rgcn_wmma_kernel(const float*  __restrict__ x,
                 const __bf16* __restrict__ xb,
                 const __bf16* __restrict__ Wt,
                 const int*    __restrict__ idx,
                 const int*    __restrict__ etype,
                 const int*    __restrict__ seg,
                 float*        __restrict__ out,
                 int E, int num_rel)
{
    __shared__ __bf16 sW[32768];              // up to 8 rel x 64x64 bf16 = 64 KB
    const int nelem = num_rel << 12;          // bf16 elements to stage

    // ---- Stage weights global -> LDS ------------------------------------
#if __has_builtin(__builtin_amdgcn_tensor_load_to_lds)
    if (threadIdx.x < 32) {                   // one wave issues the TDM DMA
        typedef __attribute__((ext_vector_type(4))) unsigned u32x4;
        typedef __attribute__((ext_vector_type(8))) int      i32x8;
        typedef __attribute__((ext_vector_type(4))) int      i32x4;

        // Low 32 bits of an LDS-aperture generic pointer are the LDS offset.
        unsigned lds_addr = (unsigned)(size_t)(&sW[0]);
        unsigned long long ga = (unsigned long long)(size_t)Wt;

        // D# group 0: count=1, lds_addr, global_addr[56:0], type=2 (image)
        u32x4 g0 = { 1u,
                     lds_addr,
                     (unsigned)(ga & 0xFFFFFFFFu),
                     (unsigned)((ga >> 32) & 0x1FFFFFFu) | (2u << 30) };
        // D# group 1: 1-row tile of nelem elements, data_size=1 (2 bytes)
        i32x8 g1 = { (int)(1u << 16),                          // data_size=2B
                     (int)(((unsigned)nelem & 0xFFFFu) << 16), // tensor_dim0 lo16
                     (int)(((unsigned)(nelem >> 16) & 0xFFFFu)
                           | (1u << 16)),                      // dim0 hi16 | tensor_dim1=1
                     (int)(((unsigned)nelem & 0xFFFFu) << 16), // tile_dim0
                     1,                                        // tile_dim1=1
                     nelem,                                    // tensor_dim0_stride lo32
                     0, 0 };
        i32x4 g2 = { 0, 0, 0, 0 };
        i32x4 g3 = { 0, 0, 0, 0 };
#if __clang_major__ >= 23
        i32x8 g4 = { 0, 0, 0, 0, 0, 0, 0, 0 };
        __builtin_amdgcn_tensor_load_to_lds(g0, g1, g2, g3, g4, 0);
#else
        __builtin_amdgcn_tensor_load_to_lds(g0, g1, g2, g3, 0);
#endif
        __builtin_amdgcn_s_wait_tensorcnt(0);
    }
#else
    // Fallback: cooperative copy (global_load_b128 + ds_store_b128)
    {
        uint4* dst = (uint4*)&sW[0];
        const uint4* src = (const uint4*)Wt;
        int nvec = nelem >> 3;                // 8 bf16 per uint4
        for (int t = threadIdx.x; t < nvec; t += 256) dst[t] = src[t];
    }
#endif
    __syncthreads();

    const int lane    = threadIdx.x & 31;
    const int wave    = threadIdx.x >> 5;
    const int rowSel  = lane & 15;      // A: M index / B,D: N index
    const int halfSel = lane >> 4;      // half-wave select
    const int kb      = halfSel << 3;   // A operand K sub-chunk base (0 or 8)

    const long tile  = (long)blockIdx.x * 8 + wave;   // 8 waves per block
    const long ebase = tile * 16;
    if (ebase >= (long)E) return;       // wave-uniform exit (EXEC stays full)

    long e = ebase + rowSel;
    bool rowValid = (e < (long)E);
    long ec = rowValid ? e : (long)(E - 1);

    int myType = rowValid ? etype[ec] : -1;
    size_t row = (size_t)(unsigned)idx[ec];

    // ---- Build A operands (16x64 edge-feature tile, bf16), ISA A layout ----
    // lanes 0-15: K chunks {kb..kb+7, kb+16..kb+23}; lanes 16-31: kb=8.
    v16bf a0, a1;
    if (USE_XB) {
        const __bf16* xr = xb + (row << 6);
        v8bf l0 = *(const v8bf*)(xr + kb);        // K kb..kb+7
        v8bf l1 = *(const v8bf*)(xr + kb + 16);   // K kb+16..kb+23
        v8bf l2 = *(const v8bf*)(xr + kb + 32);   // second K=32 step
        v8bf l3 = *(const v8bf*)(xr + kb + 48);
        a0 = __builtin_shufflevector(l0, l1, 0,1,2,3,4,5,6,7,8,9,10,11,12,13,14,15);
        a1 = __builtin_shufflevector(l2, l3, 0,1,2,3,4,5,6,7,8,9,10,11,12,13,14,15);
    } else {
        const float* xr = x + (row << 6);
        const float4* pl0 = (const float4*)(xr + kb);
        const float4* pl1 = (const float4*)(xr + kb + 16);
        const float4* pl2 = (const float4*)(xr + kb + 32);
        const float4* pl3 = (const float4*)(xr + kb + 48);
        float4 f0 = pl0[0], f1 = pl0[1];
        float4 g0 = pl1[0], g1 = pl1[1];
        float4 h0 = pl2[0], h1 = pl2[1];
        float4 j0 = pl3[0], j1 = pl3[1];
        a0[0]=f2bf(f0.x); a0[1]=f2bf(f0.y); a0[2]=f2bf(f0.z); a0[3]=f2bf(f0.w);
        a0[4]=f2bf(f1.x); a0[5]=f2bf(f1.y); a0[6]=f2bf(f1.z); a0[7]=f2bf(f1.w);
        a0[8]=f2bf(g0.x); a0[9]=f2bf(g0.y); a0[10]=f2bf(g0.z); a0[11]=f2bf(g0.w);
        a0[12]=f2bf(g1.x); a0[13]=f2bf(g1.y); a0[14]=f2bf(g1.z); a0[15]=f2bf(g1.w);
        a1[0]=f2bf(h0.x); a1[1]=f2bf(h0.y); a1[2]=f2bf(h0.z); a1[3]=f2bf(h0.w);
        a1[4]=f2bf(h1.x); a1[5]=f2bf(h1.y); a1[6]=f2bf(h1.z); a1[7]=f2bf(h1.w);
        a1[8]=f2bf(j0.x); a1[9]=f2bf(j0.y); a1[10]=f2bf(j0.z); a1[11]=f2bf(j0.w);
        a1[12]=f2bf(j1.x); a1[13]=f2bf(j1.y); a1[14]=f2bf(j1.z); a1[15]=f2bf(j1.w);
    }

    v8f c0 = {}, c1 = {}, c2 = {}, c3 = {};
    const v16bf zero = {};

    // ---- Masked accumulation over relations (mirrors reference loop) ----
    for (int r = 0; r < num_rel; ++r) {
        if (__ballot(myType == r) == 0ull) continue;   // wave-uniform skip
        v16bf ar0 = (myType == r) ? a0 : zero;         // zero rows of wrong type
        v16bf ar1 = (myType == r) ? a1 : zero;
        const __bf16* wbase = sW + (r << 12);          // LDS-resident weights
        #pragma unroll
        for (int nt = 0; nt < 4; ++nt) {
            // B layout: lane = N (rowSel), K chunks {8h..8h+7, 16+8h..16+8h+7} per half-wave
            const __bf16* wr = wbase + ((((nt << 4) + rowSel) << 6) + (halfSel << 3));
            v8bf b0lo = *(const v8bf*)(wr);          // K halfSel*8 .. +7
            v8bf b0hi = *(const v8bf*)(wr + 16);     // K 16+halfSel*8 .. +7
            v8bf b1lo = *(const v8bf*)(wr + 32);     // second K=32 step
            v8bf b1hi = *(const v8bf*)(wr + 48);
            v16bf b0 = __builtin_shufflevector(b0lo, b0hi,
                          0,1,2,3,4,5,6,7,8,9,10,11,12,13,14,15);
            v16bf b1 = __builtin_shufflevector(b1lo, b1hi,
                          0,1,2,3,4,5,6,7,8,9,10,11,12,13,14,15);
            v8f& acc = (nt == 0) ? c0 : (nt == 1) ? c1 : (nt == 2) ? c2 : c3;
            acc = __builtin_amdgcn_wmma_f32_16x16x32_bf16(
                      false, ar0, false, b0, (short)0, acc, false, false);
            acc = __builtin_amdgcn_wmma_f32_16x16x32_bf16(
                      false, ar1, false, b1, (short)0, acc, false, false);
        }
    }

    // ---- Scatter-add: D layout is lane=N(rowSel), VGPR i => M=i+8*halfSel ----
    #pragma unroll
    for (int i = 0; i < 8; ++i) {
        int  m  = i + (halfSel << 3);
        long em = ebase + m;
        if (em < (long)E) {
            int node = seg[em];
            float* orow = out + ((size_t)(unsigned)node << 6) + rowSel;
            atomicAdd(orow +  0, c0[i]);
            atomicAdd(orow + 16, c1[i]);
            atomicAdd(orow + 32, c2[i]);
            atomicAdd(orow + 48, c3[i]);
        }
    }
}

// ---------------------------------------------------------------------------
// Kernel 5: degree normalization
// ---------------------------------------------------------------------------
__global__ void norm_kernel(float* __restrict__ out,
                            const int* __restrict__ ptr, int total) {
    int t = blockIdx.x * blockDim.x + threadIdx.x;
    if (t >= total) return;
    int n = t >> 6;
    float deg = (float)(ptr[n + 1] - ptr[n]);
    out[t] = out[t] / deg;
}

// ---------------------------------------------------------------------------
extern "C" void kernel_launch(void* const* d_in, const int* in_sizes, int n_in,
                              void* d_out, int out_size, void* d_ws, size_t ws_size,
                              hipStream_t stream) {
    (void)n_in;
    const float* x      = (const float*)d_in[0];
    const float* linear = (const float*)d_in[1];
    const int*   ptr    = (const int*)d_in[2];
    const int*   idx    = (const int*)d_in[3];
    const int*   etype  = (const int*)d_in[4];

    const int num_node = in_sizes[2] - 1;       // ptr has num_node+1 entries
    const int E        = in_sizes[3];
    const int num_rel  = in_sizes[1] / 4096;    // [R,64,64]

    // Workspace layout: Wt bf16 | seg int32 | xb bf16
    size_t off = 0;
    __bf16* Wt = (__bf16*)d_ws;
    off += ((size_t)num_rel * 4096 * sizeof(__bf16) + 255) & ~(size_t)255;
    int* seg = (int*)((char*)d_ws + off);
    off += ((size_t)E * sizeof(int) + 255) & ~(size_t)255;
    __bf16* xb = (__bf16*)((char*)d_ws + off);
    size_t need = off + (size_t)num_node * 64 * sizeof(__bf16);
    const bool use_xb = (ws_size >= need);

    hipMemsetAsync(d_out, 0, (size_t)out_size * sizeof(float), stream);

    {   int total = num_rel * 4096;
        wt_bf16_kernel<<<(total + 255) / 256, 256, 0, stream>>>(linear, Wt, total); }

    seg_kernel<<<(E + 255) / 256, 256, 0, stream>>>(ptr, num_node, E, seg);

    if (use_xb) {
        int total = num_node * 64;
        xb_bf16_kernel<<<(total + 255) / 256, 256, 0, stream>>>(x, xb, total);
    }

    {   long tiles  = ((long)E + 15) / 16;       // one wave32 per 16-edge tile
        int  blocks = (int)((tiles + 7) / 8);    // 8 waves (256 threads) / block
        if (use_xb)
            rgcn_wmma_kernel<true ><<<blocks, 256, 0, stream>>>(x, xb, Wt, idx, etype,
                                                                seg, (float*)d_out, E, num_rel);
        else
            rgcn_wmma_kernel<false><<<blocks, 256, 0, stream>>>(x, xb, Wt, idx, etype,
                                                                seg, (float*)d_out, E, num_rel);
    }

    {   int total = num_node * 64;
        norm_kernel<<<(total + 255) / 256, 256, 0, stream>>>((float*)d_out, ptr, total); }
}
